// GoalTokenArgModel_53180285059837
// MI455X (gfx1250) — matrix operand order repaired and checked by reading, not compiled, not run on hardware
//
#include <hip/hip_runtime.h>
#include <hip/hip_bf16.h>

// ============================================================================
// GRU encoder + MLP head for MI455X (gfx1250, wave32, WMMA 16x16x32 f16).
//
//   setup:  convert weights to f16, xemb = f16(relu(token_emb)), h0 gather
//   per t:  gates_kernel  : GI = x_t*W_ih^T + b_ih ; GH = h*W_hh^T + b_hh
//           gru_update    : r,z,n gates -> h (fp32 + f16) + hs16[t] = relu(h)
//   EOS:    same two kernels with broadcast x = xemb[1], slot 256
//   head:   mlp_gemm(hs16,w0)->V ; mlp_gemm(V,w1)->U(=hs16 reuse) ;
//           mlp_out: per-row dot with w2 -> out[b][col]
//
// Round-4 change (gates kernel only): explicit double-buffered software
// pipeline with __builtin_amdgcn_sched_barrier(0) fences so slab i+1's
// global loads are issued before slab i's WMMA chain (partial s_wait_loadcnt
// instead of full drains), plus a 64x64 macro tile -> 96 blocks for more
// co-resident waves on the latency-critical sequential step.
// ============================================================================

typedef __attribute__((ext_vector_type(16))) _Float16 v16h;
typedef __attribute__((ext_vector_type(8)))  _Float16 v8h;
typedef __attribute__((ext_vector_type(8)))  float    v8f;

#define B_    256
#define L_    256
#define H_    512
#define G3H   1536
#define NSLOT 257
#define NROWS (NSLOT * B_)   // 65792 rows through the MLP head

// ---------------------------------------------------------------------------
// WMMA fragment helpers (layouts per CDNA5 ISA §7.12.2).
// base = kernel-arg pointer (SGPR pair); off = 32-bit per-lane element index.
// ---------------------------------------------------------------------------

// A: 16(M) x 32(K); lane holds row m = lane&15, halfsel = lane>>4 selects
// K {0..7,16..23} vs {8..15,24..31}.  off = row*ld + halfsel*8 + kk.
__device__ __forceinline__ v16h load_a_frag(const _Float16* __restrict__ base,
                                            int off) {
  v8h lo = *(const v8h*)(base + off);        // K rel = halfsel*8 + [0..7]
  v8h hi = *(const v8h*)(base + off + 16);   // K rel = 16 + halfsel*8 + [0..7]
  v16h r;
#pragma unroll
  for (int i = 0; i < 8; ++i) { r[i] = lo[i]; r[i + 8] = hi[i]; }
  return r;
}

// B: 32(K) x 16(N) from W row-major [N][K] (B = W^T, same K order as A).
// off = (nBase + (lane&15))*ld + (lane>>4)*8 + kk.
__device__ __forceinline__ v16h load_b_frag(const _Float16* __restrict__ base,
                                            int off) {
  v8h lo = *(const v8h*)(base + off);
  v8h hi = *(const v8h*)(base + off + 16);
  v16h r;
#pragma unroll
  for (int i = 0; i < 8; ++i) { r[i] = lo[i]; r[i + 8] = hi[i]; }
  return r;
}

// ---------------------------------------------------------------------------
// Setup kernels
// ---------------------------------------------------------------------------
__global__ void cvt_f16_kernel(const float* __restrict__ src,
                               _Float16* __restrict__ dst, int n) {
  int i = blockIdx.x * blockDim.x + threadIdx.x;
  if (i < n) dst[i] = (_Float16)src[i];
}

__global__ void relu_cvt_f16_kernel(const float* __restrict__ src,
                                    _Float16* __restrict__ dst, int n) {
  int i = blockIdx.x * blockDim.x + threadIdx.x;
  if (i < n) dst[i] = (_Float16)fmaxf(src[i], 0.0f);
}

__global__ void init_h_kernel(const float* __restrict__ stem_emb,
                              const int* __restrict__ stem,
                              float* __restrict__ h,
                              _Float16* __restrict__ h16) {
  int idx = blockIdx.x * blockDim.x + threadIdx.x;   // 0..B*H-1
  int b = idx >> 9;
  int k = idx & (H_ - 1);
  float v = stem_emb[(size_t)stem[b] * H_ + k];
  h[idx]   = v;
  h16[idx] = (_Float16)v;
}

// ---------------------------------------------------------------------------
// Per-timestep gates GEMM:  GI/GH [256 x 1536], K = 512.
// Block = 256 thr = 8 waves -> 64(M) x 64(N) macro tile; wave does 16 x 32
// with TWO accumulator sets (GI from x_t, GH from h).  Grid = (4, 24).
// Double-buffered K pipeline: slab i+1 loads issued before slab i WMMAs.
// t >= 0: x_t row b = xemb[goal[b][t]];  t < 0: EOS broadcast row xemb[1].
// ---------------------------------------------------------------------------
__global__ void __launch_bounds__(256, 1)
gates_kernel(const _Float16* __restrict__ xemb,
             const _Float16* __restrict__ h16,
             const _Float16* __restrict__ w_ih16,
             const _Float16* __restrict__ w_hh16,
             const float* __restrict__ b_ih,
             const float* __restrict__ b_hh,
             const int* __restrict__ goal,
             int t,
             float* __restrict__ GI,
             float* __restrict__ GH) {
  const int lane    = threadIdx.x & 31;
  const int wave    = threadIdx.x >> 5;
  const int mBase   = blockIdx.x * 64 + (wave >> 1) * 16;
  const int nBase   = blockIdx.y * 64 + (wave & 1) * 32;
  const int halfsel = lane >> 4;
  const int n15     = lane & 15;
  const int mrow    = mBase + n15;

  const int tok = (t >= 0) ? goal[mrow * L_ + t] : 1;  // EOS_TOKEN = 1

  // 32-bit per-lane element offsets (SADDR + voffset addressing).
  const int xoff = tok  * H_ + halfsel * 8;
  const int hoff = mrow * H_ + halfsel * 8;
  const int boff = (nBase + n15) * H_ + halfsel * 8;   // same for both weights

  const v8f vzero = {0.f, 0.f, 0.f, 0.f, 0.f, 0.f, 0.f, 0.f};
  v8f accI[2] = {vzero, vzero};
  v8f accH[2] = {vzero, vzero};

  // Double buffers (fully unrolled loop -> all in registers).
  v16h bi[2][2], bh[2][2], axb[2], ahb[2];

  // Prologue: slab 0.
  bi[0][0] = load_b_frag(w_ih16, boff);
  bi[0][1] = load_b_frag(w_ih16, boff + 16 * H_);
  bh[0][0] = load_b_frag(w_hh16, boff);
  bh[0][1] = load_b_frag(w_hh16, boff + 16 * H_);
  axb[0]   = load_a_frag(xemb, xoff);
  ahb[0]   = load_a_frag(h16,  hoff);

#pragma unroll
  for (int it = 0; it < 16; ++it) {
    const int cur = it & 1;
    const int nxt = cur ^ 1;
    const int kn  = (it + 1) * 32;
    if (it < 15) {
      // Prefetch slab it+1 (no consumer until next iteration).
      bi[nxt][0] = load_b_frag(w_ih16, boff + kn);
      bi[nxt][1] = load_b_frag(w_ih16, boff + kn + 16 * H_);
      bh[nxt][0] = load_b_frag(w_hh16, boff + kn);
      bh[nxt][1] = load_b_frag(w_hh16, boff + kn + 16 * H_);
      axb[nxt]   = load_a_frag(xemb, xoff + kn);
      ahb[nxt]   = load_a_frag(h16,  hoff + kn);
    }
    __builtin_amdgcn_sched_barrier(0);   // keep prefetch above the math
    accI[0] = __builtin_amdgcn_wmma_f32_16x16x32_f16(
        false, axb[cur], false, bi[cur][0], (short)0, accI[0], false, false);
    accI[1] = __builtin_amdgcn_wmma_f32_16x16x32_f16(
        false, axb[cur], false, bi[cur][1], (short)0, accI[1], false, false);
    accH[0] = __builtin_amdgcn_wmma_f32_16x16x32_f16(
        false, ahb[cur], false, bh[cur][0], (short)0, accH[0], false, false);
    accH[1] = __builtin_amdgcn_wmma_f32_16x16x32_f16(
        false, ahb[cur], false, bh[cur][1], (short)0, accH[1], false, false);
    __builtin_amdgcn_sched_barrier(0);
  }

#pragma unroll
  for (int j = 0; j < 2; ++j) {
    int n = nBase + 16 * j + n15;
    float bi_s = b_ih[n], bh_s = b_hh[n];
#pragma unroll
    for (int r = 0; r < 8; ++r) {
      int m = mBase + r + 8 * halfsel;
      GI[m * G3H + n] = accI[j][r] + bi_s;
      GH[m * G3H + n] = accH[j][r] + bh_s;
    }
  }
}

// ---------------------------------------------------------------------------
// Elementwise GRU gate combine: h_new = (1-z)*n + z*h  (fp32 math).
// Writes h (fp32), h16 (f16, next step's A operand), hs16 slot = relu(h).
// ---------------------------------------------------------------------------
__global__ void gru_update_kernel(const float* __restrict__ GI,
                                  const float* __restrict__ GH,
                                  float* __restrict__ h,
                                  _Float16* __restrict__ h16,
                                  _Float16* __restrict__ hs16_slot) {
  int idx = blockIdx.x * blockDim.x + threadIdx.x;   // 0..B*H-1
  int b = idx >> 9;
  int j = idx & (H_ - 1);
  int gb = b * G3H;
  float gir = GI[gb + j], giz = GI[gb + 512 + j], gin = GI[gb + 1024 + j];
  float ghr = GH[gb + j], ghz = GH[gb + 512 + j], ghn = GH[gb + 1024 + j];
  float r = 1.f / (1.f + __expf(-(gir + ghr)));
  float z = 1.f / (1.f + __expf(-(giz + ghz)));
  float n = tanhf(gin + r * ghn);
  float hn = (1.f - z) * n + z * h[idx];
  h[idx]         = hn;
  h16[idx]       = (_Float16)hn;
  hs16_slot[idx] = (_Float16)fmaxf(hn, 0.f);
}

// ---------------------------------------------------------------------------
// Batched MLP hidden layer: C[M x 512] = f16(relu(A[M x 512] * W^T + bias)).
// Block = 8 waves -> 64 x 128 tile; wave does 16 x 64.  Grid = (M/64, 4).
// Occupancy (4112 blocks) hides latency here; keep the simple structure.
// ---------------------------------------------------------------------------
__global__ void __launch_bounds__(256, 2)
mlp_gemm_kernel(const _Float16* __restrict__ A,
                const _Float16* __restrict__ W,
                const float* __restrict__ bias,
                _Float16* __restrict__ C) {
  const int lane    = threadIdx.x & 31;
  const int wave    = threadIdx.x >> 5;
  const int mBase   = blockIdx.x * 64 + (wave >> 1) * 16;
  const int nBase   = blockIdx.y * 128 + (wave & 1) * 64;
  const int halfsel = lane >> 4;
  const int n15     = lane & 15;

  const int aoff = (mBase + n15) * H_ + halfsel * 8;   // < 65792*512 < 2^26
  const int boff = (nBase + n15) * H_ + halfsel * 8;

  const v8f vzero = {0.f, 0.f, 0.f, 0.f, 0.f, 0.f, 0.f, 0.f};
  v8f acc[4];
#pragma unroll
  for (int j = 0; j < 4; ++j) acc[j] = vzero;

#pragma unroll 4
  for (int kk = 0; kk < H_; kk += 32) {
    v16h b[4];
#pragma unroll
    for (int j = 0; j < 4; ++j)
      b[j] = load_b_frag(W, boff + kk + j * (16 * H_));
    v16h a = load_a_frag(A, aoff + kk);
#pragma unroll
    for (int j = 0; j < 4; ++j)
      acc[j] = __builtin_amdgcn_wmma_f32_16x16x32_f16(
          false, a, false, b[j], (short)0, acc[j], false, false);
  }

#pragma unroll
  for (int j = 0; j < 4; ++j) {
    int n = nBase + 16 * j + n15;
    float bv = bias[n];
#pragma unroll
    for (int r = 0; r < 8; ++r) {
      int m = mBase + r + 8 * halfsel;
      C[m * H_ + n] = (_Float16)fmaxf(acc[j][r] + bv, 0.f);
    }
  }
}

// ---------------------------------------------------------------------------
// Output layer: out row dot w2 (+b2), one wave per row, wave32 shfl reduce.
// Row -> (slot, b); slot 256 (EOS) -> col 0, slot t -> col t+1.
// ---------------------------------------------------------------------------
__global__ void mlp_out_kernel(const _Float16* __restrict__ U,
                               const _Float16* __restrict__ w2,
                               const float* __restrict__ b2,
                               float* __restrict__ out) {
  const int lane = threadIdx.x & 31;
  const int wave = threadIdx.x >> 5;
  const int row  = blockIdx.x * 8 + wave;

  const int uoff = row * H_ + lane * 16;   // 16 f16 per lane
  v8h u0 = *(const v8h*)(U + uoff);
  v8h u1 = *(const v8h*)(U + uoff + 8);
  v8h w0 = *(const v8h*)(w2 + lane * 16);
  v8h w1 = *(const v8h*)(w2 + lane * 16 + 8);
  float acc = 0.f;
#pragma unroll
  for (int i = 0; i < 8; ++i)
    acc += (float)u0[i] * (float)w0[i] + (float)u1[i] * (float)w1[i];
#pragma unroll
  for (int off = 16; off > 0; off >>= 1)
    acc += __shfl_xor(acc, off, 32);

  if (lane == 0) {
    int slot = row >> 8;        // / 256
    int b    = row & 255;
    int col  = (slot == 256) ? 0 : (slot + 1);
    out[b * NSLOT + col] = acc + b2[0];
  }
}

// ---------------------------------------------------------------------------
// Host side
// ---------------------------------------------------------------------------
extern "C" void kernel_launch(void* const* d_in, const int* in_sizes, int n_in,
                              void* d_out, int out_size, void* d_ws, size_t ws_size,
                              hipStream_t stream) {
  const int*   stem_batch = (const int*)  d_in[0];
  const int*   goal_batch = (const int*)  d_in[1];
  const float* stem_emb   = (const float*)d_in[2];
  const float* token_emb  = (const float*)d_in[3];
  const float* w_ih       = (const float*)d_in[4];
  const float* w_hh       = (const float*)d_in[5];
  const float* b_ih       = (const float*)d_in[6];
  const float* b_hh       = (const float*)d_in[7];
  const float* w0         = (const float*)d_in[8];
  const float* b0         = (const float*)d_in[9];
  const float* w1         = (const float*)d_in[10];
  const float* b1         = (const float*)d_in[11];
  const float* w2         = (const float*)d_in[12];
  const float* b2         = (const float*)d_in[13];
  float* out = (float*)d_out;

  // Workspace carve-up (all 256B aligned by construction). ~137 MiB total.
  char* ws = (char*)d_ws;
  size_t off = 0;
  auto carve = [&](size_t bytes) { char* p = ws + off; off += (bytes + 255) & ~(size_t)255; return p; };
  _Float16* w_ih16 = (_Float16*)carve((size_t)G3H * H_ * 2);   // 1.5 MB
  _Float16* w_hh16 = (_Float16*)carve((size_t)G3H * H_ * 2);   // 1.5 MB
  _Float16* w0_16  = (_Float16*)carve((size_t)H_ * H_ * 2);    // 0.5 MB
  _Float16* w1_16  = (_Float16*)carve((size_t)H_ * H_ * 2);    // 0.5 MB
  _Float16* w2_16  = (_Float16*)carve((size_t)H_ * 2);
  _Float16* xemb16 = (_Float16*)carve((size_t)128 * H_ * 2);   // relu(token_emb)
  float*    h_f32  = (float*)   carve((size_t)B_ * H_ * 4);
  _Float16* h16    = (_Float16*)carve((size_t)B_ * H_ * 2);
  float*    GI     = (float*)   carve((size_t)B_ * G3H * 4);   // 1.5 MB
  float*    GH     = (float*)   carve((size_t)B_ * G3H * 4);   // 1.5 MB
  _Float16* hs16   = (_Float16*)carve((size_t)NROWS * H_ * 2); // 67 MB (also reused as U)
  _Float16* V16    = (_Float16*)carve((size_t)NROWS * H_ * 2); // 67 MB
  (void)ws_size; (void)in_sizes; (void)n_in; (void)out_size;

  // --- setup: weight conversion, relu'd token embedding, h0 gather ---
  cvt_f16_kernel<<<(G3H * H_) / 256, 256, 0, stream>>>(w_ih, w_ih16, G3H * H_);
  cvt_f16_kernel<<<(G3H * H_) / 256, 256, 0, stream>>>(w_hh, w_hh16, G3H * H_);
  cvt_f16_kernel<<<(H_ * H_) / 256, 256, 0, stream>>>(w0, w0_16, H_ * H_);
  cvt_f16_kernel<<<(H_ * H_) / 256, 256, 0, stream>>>(w1, w1_16, H_ * H_);
  cvt_f16_kernel<<<2, 256, 0, stream>>>(w2, w2_16, H_);
  relu_cvt_f16_kernel<<<(128 * H_) / 256, 256, 0, stream>>>(token_emb, xemb16, 128 * H_);
  init_h_kernel<<<(B_ * H_) / 256, 256, 0, stream>>>(stem_emb, stem_batch, h_f32, h16);

  // --- recurrence: 256 timesteps + EOS step ---
  dim3 ggrid(B_ / 64, G3H / 64);   // (4, 24) = 96 blocks
  for (int t = 0; t < L_; ++t) {
    gates_kernel<<<ggrid, 256, 0, stream>>>(xemb16, h16, w_ih16, w_hh16,
                                            b_ih, b_hh, goal_batch, t, GI, GH);
    gru_update_kernel<<<(B_ * H_) / 256, 256, 0, stream>>>(
        GI, GH, h_f32, h16, hs16 + (size_t)t * B_ * H_);
  }
  gates_kernel<<<ggrid, 256, 0, stream>>>(xemb16, h16, w_ih16, w_hh16,
                                          b_ih, b_hh, goal_batch, -1, GI, GH);
  gru_update_kernel<<<(B_ * H_) / 256, 256, 0, stream>>>(
      GI, GH, h_f32, h16, hs16 + (size_t)L_ * B_ * H_);   // slot 256 = EOS

  // --- batched MLP head over all 257*256 rows ---
  dim3 mgrid(NROWS / 64, H_ / 128);  // (1028, 4)
  mlp_gemm_kernel<<<mgrid, 256, 0, stream>>>(hs16, w0_16, b0, V16);
  mlp_gemm_kernel<<<mgrid, 256, 0, stream>>>(V16, w1_16, b1, hs16); // U reuses hs16
  mlp_out_kernel<<<NROWS / 8, 256, 0, stream>>>(hs16, w2_16, b2, out);
}
